// HeterogeneousKANLayer_83425444757981
// MI455X (gfx1250) — matrix operand order repaired and checked by reading, not compiled, not run on hardware
//
#include <hip/hip_runtime.h>
#include <hip/hip_bf16.h>

typedef __attribute__((ext_vector_type(16))) _Float16 v16h;
typedef __attribute__((ext_vector_type(8)))  _Float16 v8h;
typedef __attribute__((ext_vector_type(8)))  float    v8f;

#define BATCH   4096
#define IN_DIM  512
#define OUT_DIM 512
#define NCENT   10
#define KPACK   11                 // 10 RBF centers + 1 silu(base) channel
#define KTOT    (IN_DIM * KPACK)   // 5632, divisible by BK=32

#define BM 128
#define BN 128
#define BK 32
#define LDA 56    // f16 row stride of As (16B aligned rows, bank-staggered)
#define LDB 136   // f16 row stride of Bs (16B aligned rows, bank-staggered)

// ---------------------------------------------------------------------------
// Kernel 1: pack activations A[b, i*KPACK + c] (f16)
//   c in [0,10): exp(-((tanh(x)-center_c)/width)^2), width = 2/9
//   c == 10   : silu(tanh(x))
// ---------------------------------------------------------------------------
__global__ void kan_pack_act(const float* __restrict__ x, _Float16* __restrict__ A)
{
    int idx = blockIdx.x * blockDim.x + threadIdx.x;   // idx = b*IN_DIM + i
    if (idx >= BATCH * IN_DIM) return;

    float xv = x[idx];
    float e2 = __expf(2.0f * xv);          // tanh = 1 - 2/(e^{2x}+1)
    float t  = 1.0f - 2.0f / (e2 + 1.0f);

    _Float16* dst = A + (size_t)idx * KPACK;
    const float inv_w = 4.5f;              // 1/width, width = 2/(NCENT-1)
#pragma unroll
    for (int c = 0; c < NCENT; ++c) {
        float cc = -1.0f + (2.0f / 9.0f) * (float)c;
        float z  = (t - cc) * inv_w;
        dst[c] = (_Float16)__expf(-z * z);
    }
    float sig = 1.0f / (1.0f + __expf(-t));
    dst[NCENT] = (_Float16)(t * sig);      // silu(tanh(x))
}

// ---------------------------------------------------------------------------
// Kernel 2: pack weights W[k, o] (f16), k = i*KPACK + c
// ---------------------------------------------------------------------------
__global__ void kan_pack_w(const float* __restrict__ coef,
                           const float* __restrict__ scale_base,
                           const float* __restrict__ scale_sp,
                           _Float16* __restrict__ W)
{
    int idx = blockIdx.x * blockDim.x + threadIdx.x;   // idx = k*OUT_DIM + o
    if (idx >= KTOT * OUT_DIM) return;
    int k = idx >> 9;
    int o = idx & (OUT_DIM - 1);
    int i = k / KPACK;
    int c = k - i * KPACK;
    float w;
    if (c < NCENT)
        w = coef[((size_t)i * OUT_DIM + o) * NCENT + c] * scale_sp[(size_t)o * IN_DIM + i];
    else
        w = scale_base[(size_t)o * IN_DIM + i];
    W[idx] = (_Float16)w;
}

// ---------------------------------------------------------------------------
// Kernel 3: out = A[4096,KTOT] x W[KTOT,512], f32 accumulate.
// 256 threads = 8 wave32s (2Mx4N); 128x128 block tile; BK=32.
// Double-buffered LDS fed by GLOBAL_LOAD_ASYNC_TO_LDS_B128 (ASYNCcnt),
// one workgroup barrier per K-step.
// ---------------------------------------------------------------------------
__global__ __launch_bounds__(256)
void kan_wmma_gemm(const _Float16* __restrict__ A,
                   const _Float16* __restrict__ W,
                   float* __restrict__ out)
{
    __shared__ __align__(16) _Float16 As[2][BM * LDA];
    __shared__ __align__(16) _Float16 Bs[2][BK * LDB];

    const int tid  = threadIdx.x;
    const int lane = tid & 31;
    const int wave = tid >> 5;
    const int wm   = wave & 1;     // 0..1 -> 64-row slab
    const int wn   = wave >> 1;    // 0..3 -> 32-col slab
    const int m0   = blockIdx.y * BM;
    const int n0   = blockIdx.x * BN;

    // staging maps: A tile 128x32 (2 threads/row), B tile 32x128 (8 threads/row)
    const int arow = tid >> 1, ahalf = tid & 1;
    const int brow = tid >> 3, bseg  = tid & 7;

    const _Float16* gA = A + (size_t)(m0 + arow) * KTOT + ahalf * 16;
    const _Float16* gB = W + (size_t)brow * OUT_DIM + n0 + bseg * 16;

    // Issue this thread's share of one K-tile (4 async b128) into buffer `buf`.
    // NOTE: IOFFSET is added to BOTH the LDS and global addresses (ISA §10),
    // so offset:16 fetches the second 16B chunk of each 32B segment.
    auto issue = [&](int k0, int buf) {
        unsigned la = (unsigned)(size_t)(&As[buf][arow * LDA + ahalf * 16]); // low 32b = LDS offset
        unsigned lb = (unsigned)(size_t)(&Bs[buf][brow * LDB + bseg * 16]);
        unsigned long long ga = (unsigned long long)(size_t)(gA + k0);
        unsigned long long gb = (unsigned long long)(size_t)(gB + (size_t)k0 * OUT_DIM);
        asm volatile("global_load_async_to_lds_b128 %0, %1, off"
                     :: "v"(la), "v"(ga) : "memory");
        asm volatile("global_load_async_to_lds_b128 %0, %1, off offset:16"
                     :: "v"(la), "v"(ga) : "memory");
        asm volatile("global_load_async_to_lds_b128 %0, %1, off"
                     :: "v"(lb), "v"(gb) : "memory");
        asm volatile("global_load_async_to_lds_b128 %0, %1, off offset:16"
                     :: "v"(lb), "v"(gb) : "memory");
    };

    v8f acc[4][2];
#pragma unroll
    for (int mf = 0; mf < 4; ++mf)
#pragma unroll
        for (int nf = 0; nf < 2; ++nf)
            acc[mf][nf] = (v8f){0.f, 0.f, 0.f, 0.f, 0.f, 0.f, 0.f, 0.f};

    // Per-lane fragment geometry (ISA 7.12.2):
    //  A: lane<16 -> row lane,     K chunks {0..7} and {16..23}
    //     lane>=16 -> row lane-16, K chunks {8..15} and {24..31}
    //  B: lane l -> col lane&15, same K chunks as A (stride LDB).
    const int aRow  = wm * 64 + (lane & 15);
    const int aKoff = (lane < 16) ? 0 : 8;
    const int NT    = KTOT / BK;   // 176

    issue(0, 0);

    for (int t = 0; t < NT; ++t) {
        const int cur = t & 1;

        // my async writes into `cur` done; barrier publishes them to all waves
        // and proves everyone finished reading `cur^1` last iteration.
        asm volatile("s_wait_asynccnt 0x0" ::: "memory");
        __syncthreads();

        if (t + 1 < NT) issue((t + 1) * BK, cur ^ 1);   // prefetch next tile

        // ---- load fragments from LDS buffer `cur` ----
        v16h afrag[4];
#pragma unroll
        for (int mf = 0; mf < 4; ++mf) {
            const _Float16* p = &As[cur][(aRow + mf * 16) * LDA];
            v8h lo = *(const v8h*)(p + aKoff);
            v8h hi = *(const v8h*)(p + aKoff + 16);
            afrag[mf] = __builtin_shufflevector(lo, hi, 0, 1, 2, 3, 4, 5, 6, 7,
                                                        8, 9, 10, 11, 12, 13, 14, 15);
        }
        v16h bfrag[2];
#pragma unroll
        for (int nf = 0; nf < 2; ++nf) {
            const _Float16* p = &Bs[cur][wn * 32 + nf * 16 + (lane & 15)];
#pragma unroll
            for (int s = 0; s < 16; ++s)
                bfrag[nf][s] = p[(aKoff + (s & 7) + (s >> 3) * 16) * LDB];
        }

        // ---- 8 WMMA per wave per K-step ----
#pragma unroll
        for (int mf = 0; mf < 4; ++mf)
#pragma unroll
            for (int nf = 0; nf < 2; ++nf)
                acc[mf][nf] = __builtin_amdgcn_wmma_f32_16x16x32_f16(
                    false, afrag[mf], false, bfrag[nf],
                    (short)0, acc[mf][nf], false, false);
    }

    // ---- epilogue: C/D layout -> lanes 0-15: M=r, N=lane; lanes 16-31: M=r+8 ----
#pragma unroll
    for (int mf = 0; mf < 4; ++mf) {
#pragma unroll
        for (int nf = 0; nf < 2; ++nf) {
            int col   = n0 + wn * 32 + nf * 16 + (lane & 15);
            int rbase = m0 + wm * 64 + mf * 16 + ((lane < 16) ? 0 : 8);
#pragma unroll
            for (int r = 0; r < 8; ++r)
                out[(size_t)(rbase + r) * OUT_DIM + col] = acc[mf][nf][r];
        }
    }
}

// ---------------------------------------------------------------------------
extern "C" void kernel_launch(void* const* d_in, const int* in_sizes, int n_in,
                              void* d_out, int out_size, void* d_ws, size_t ws_size,
                              hipStream_t stream)
{
    const float* x    = (const float*)d_in[0];   // [4096, 512]
    const float* coef = (const float*)d_in[1];   // [512, 512, 10]
    const float* sb   = (const float*)d_in[2];   // [512, 512] scale_base [O,I]
    const float* ssp  = (const float*)d_in[3];   // [512, 512] scale_sp   [O,I]
    float* out        = (float*)d_out;           // [4096, 512]

    // workspace: A f16 [4096, 5632] then W f16 [5632, 512]  (~52 MB total)
    _Float16* Af = (_Float16*)d_ws;
    _Float16* Wf = Af + (size_t)BATCH * KTOT;

    {
        int total = BATCH * IN_DIM;
        kan_pack_act<<<(total + 255) / 256, 256, 0, stream>>>(x, Af);
    }
    {
        int total = KTOT * OUT_DIM;
        kan_pack_w<<<(total + 255) / 256, 256, 0, stream>>>(coef, sb, ssp, Wf);
    }
    {
        dim3 grid(OUT_DIM / BN, BATCH / BM);   // 4 x 32 = 128 blocks
        kan_wmma_gemm<<<grid, 256, 0, stream>>>(Af, Wf, out);
    }
}
